// ConditionalRandomField_45732811768101
// MI455X (gfx1250) — matrix operand order repaired and checked by reading, not compiled, not run on hardware
//
#include <hip/hip_runtime.h>
#include <stdint.h>

typedef __attribute__((ext_vector_type(16))) _Float16 v16h;
typedef __attribute__((ext_vector_type(8)))  _Float16 v8h;
typedef __attribute__((ext_vector_type(8)))  float    v8f;

#define T_LEN 2048
#define BATCH 256
#define KTAG  64
#define LDSTR 72   // padded LDS row stride in halves (144B: 16B aligned, spreads banks)

#define LOG2E 1.44269504088896340736f
#define LN2   0.69314718055994530942f

// ---------------------------------------------------------------------------
// Forward algorithm: one wave (32 lanes) per 16-row batch tile.
// alpha kept in WMMA C/D layout: 4 j-tiles x v8f.
//   D layout: VGPR k, lanes 0-15 -> row k, lanes 16-31 -> row k+8; N = lane&15.
// Per step:  rowmax -> exp2 -> LDS stage -> A operands -> 8x WMMA -> m + log2.
// B operands = exp(transitions) in f16, loaded once, resident in VGPRs.
// emit/mask for step t+1 preloaded during step t's WMMA chain.
// Masked update is a branchless float blend: alpha += mf*(na - alpha).
// ---------------------------------------------------------------------------
__global__ __launch_bounds__(32) void crf_forward_wmma(
    const float* __restrict__ logits,
    const int*   __restrict__ mask,
    const float* __restrict__ transitions,
    const float* __restrict__ start_trans,
    const float* __restrict__ end_trans,
    float* __restrict__ allPath)
{
    __shared__ __align__(16) _Float16 Ecol[KTAG][LDSTR]; // Ecol[j][i] = exp(trans[i][j])
    __shared__ __align__(16) _Float16 Aexp[16][LDSTR];   // staging for A operand

    const int lane = threadIdx.x;       // 0..31
    const int r0   = blockIdx.x * 16;   // batch tile base
    const int half = lane >> 4;         // 0 | 1
    const int lcol = lane & 15;
    const int mrow = half * 8;

    // ---- build E column-major (contiguous in i) in f16 ----
    for (int idx = lane; idx < KTAG * KTAG; idx += 32) {
        int i = idx & 63, j = idx >> 6;
        Ecol[j][i] = (_Float16)__builtin_amdgcn_exp2f(transitions[i * KTAG + j] * LOG2E);
    }
    __syncthreads();

    // ---- load 8 resident B operands: jt in 0..3 (N tile), hh in 0..1 (k half) ----
    // B 32x16 f16 layout: lane holds col=lane&15, K chunks [k0..k0+7] and [k0+16..k0+23], k0 = (lane>>4)*8.
    v16h Bop[4][2];
    for (int jt = 0; jt < 4; ++jt) {
        for (int hh = 0; hh < 2; ++hh) {
            const _Float16* p = &Ecol[jt * 16 + lcol][hh * 32 + half * 8];
            v8h lo = *(const v8h*)p;
            v8h hi = *(const v8h*)(p + 16);
            v16h b;
            #pragma unroll
            for (int e = 0; e < 8; ++e) { b[e] = lo[e]; b[e + 8] = hi[e]; }
            Bop[jt][hh] = b;
        }
    }

    // ---- per-lane address bases (D layout rows) ----
    long rowoff[8], maskoff[8];
    #pragma unroll
    for (int k = 0; k < 8; ++k) {
        rowoff[k]  = (long)(r0 + mrow + k) * T_LEN * KTAG + lcol;
        maskoff[k] = (long)(r0 + mrow + k) * T_LEN;
    }

    // ---- t = 0: alpha = emit + start_trans ----
    float st[4];
    #pragma unroll
    for (int jt = 0; jt < 4; ++jt) st[jt] = start_trans[jt * 16 + lcol];

    v8f alpha[4];
    #pragma unroll
    for (int jt = 0; jt < 4; ++jt)
        #pragma unroll
        for (int k = 0; k < 8; ++k)
            alpha[jt][k] = logits[rowoff[k] + jt * 16] + st[jt];

    // ---- software-pipelined preload of emit / float-mask for t = 1 ----
    float emitc[4][8];
    float mfc[8];                        // mask as {0.0f, 1.0f}
    #pragma unroll
    for (int k = 0; k < 8; ++k) {
        mfc[k] = (mask[maskoff[k] + 1] > 0) ? 1.0f : 0.0f;
        #pragma unroll
        for (int jt = 0; jt < 4; ++jt)
            emitc[jt][k] = logits[rowoff[k] + (long)1 * KTAG + jt * 16];
    }

    // ---- sequential scan over time ----
    #pragma unroll 1
    for (int t = 1; t < T_LEN; ++t) {
        // prefetch emit stream ahead (global_prefetch_b8)
        __builtin_prefetch(&logits[rowoff[0] + (long)(t + 8) * KTAG], 0, 0);
        __builtin_prefetch(&logits[rowoff[4] + (long)(t + 8) * KTAG], 0, 0);

        // per-row max over all 64 j (4 tiles + 16-lane butterfly within half)
        float m[8];
        #pragma unroll
        for (int k = 0; k < 8; ++k) {
            float x = fmaxf(fmaxf(alpha[0][k], alpha[1][k]),
                            fmaxf(alpha[2][k], alpha[3][k]));
            x = fmaxf(x, __shfl_xor(x, 1, 16));
            x = fmaxf(x, __shfl_xor(x, 2, 16));
            x = fmaxf(x, __shfl_xor(x, 4, 16));
            x = fmaxf(x, __shfl_xor(x, 8, 16));
            m[k] = x;
        }

        // exp2((alpha - m) * log2e) -> f16 -> LDS row-major [row][i]
        #pragma unroll
        for (int jt = 0; jt < 4; ++jt)
            #pragma unroll
            for (int k = 0; k < 8; ++k)
                Aexp[mrow + k][jt * 16 + lcol] = (_Float16)
                    __builtin_amdgcn_exp2f((alpha[jt][k] - m[k]) * LOG2E);
        __syncthreads();

        // A operands. A 16x32 f16 layout: lane row = lane&15,
        // K chunks [k0..k0+7] and [k0+16..k0+23], k0 = (lane>>4)*8.
        v16h Aop[2];
        #pragma unroll
        for (int hh = 0; hh < 2; ++hh) {
            const _Float16* p0 = &Aexp[lcol][hh * 32 + half * 8];
            v8h lo = *(const v8h*)p0;
            v8h hi = *(const v8h*)(p0 + 16);
            v16h a;
            #pragma unroll
            for (int e = 0; e < 8; ++e) { a[e] = lo[e]; a[e + 8] = hi[e]; }
            Aop[hh] = a;
        }

        // 8 x v_wmma_f32_16x16x32_f16 : P[16x64] = Aexp x E
        v8f P[4];
        #pragma unroll
        for (int jt = 0; jt < 4; ++jt) {
            v8f c = {};
            c = __builtin_amdgcn_wmma_f32_16x16x32_f16(
                    false, Aop[0], false, Bop[jt][0], (short)0, c, false, false);
            c = __builtin_amdgcn_wmma_f32_16x16x32_f16(
                    false, Aop[1], false, Bop[jt][1], (short)0, c, false, false);
            P[jt] = c;
        }

        // preload emit / float-mask for step t+1 (clamped) while WMMAs in flight
        const long tn = (t + 1 < T_LEN) ? (long)(t + 1) : (long)(T_LEN - 1);
        float emitn[4][8];
        float mfn[8];
        #pragma unroll
        for (int k = 0; k < 8; ++k) {
            mfn[k] = (mask[maskoff[k] + tn] > 0) ? 1.0f : 0.0f;
            #pragma unroll
            for (int jt = 0; jt < 4; ++jt)
                emitn[jt][k] = logits[rowoff[k] + tn * KTAG + jt * 16];
        }

        // alpha' = m + ln2*log2(P) + emit, blended branchlessly by float mask:
        //   alpha = fma(mf, na - alpha, alpha)   (mf in {0,1})
        #pragma unroll
        for (int jt = 0; jt < 4; ++jt) {
            #pragma unroll
            for (int k = 0; k < 8; ++k) {
                float na = __builtin_fmaf(LN2, __builtin_amdgcn_logf(P[jt][k]), m[k])
                         + emitc[jt][k];
                alpha[jt][k] = __builtin_fmaf(mfc[k], na - alpha[jt][k], alpha[jt][k]);
            }
        }
        #pragma unroll
        for (int k = 0; k < 8; ++k) {
            mfc[k] = mfn[k];
            #pragma unroll
            for (int jt = 0; jt < 4; ++jt) emitc[jt][k] = emitn[jt][k];
        }
        __syncthreads();  // protect Aexp before next iteration's stores
    }

    // ---- all_path[b] = logsumexp_j(alpha + end_trans) ----
    float et[4];
    #pragma unroll
    for (int jt = 0; jt < 4; ++jt) et[jt] = end_trans[jt * 16 + lcol];

    #pragma unroll
    for (int k = 0; k < 8; ++k) {
        float x0 = alpha[0][k] + et[0];
        float x1 = alpha[1][k] + et[1];
        float x2 = alpha[2][k] + et[2];
        float x3 = alpha[3][k] + et[3];
        float mx = fmaxf(fmaxf(x0, x1), fmaxf(x2, x3));
        mx = fmaxf(mx, __shfl_xor(mx, 1, 16));
        mx = fmaxf(mx, __shfl_xor(mx, 2, 16));
        mx = fmaxf(mx, __shfl_xor(mx, 4, 16));
        mx = fmaxf(mx, __shfl_xor(mx, 8, 16));
        float s = __builtin_amdgcn_exp2f((x0 - mx) * LOG2E)
                + __builtin_amdgcn_exp2f((x1 - mx) * LOG2E)
                + __builtin_amdgcn_exp2f((x2 - mx) * LOG2E)
                + __builtin_amdgcn_exp2f((x3 - mx) * LOG2E);
        s += __shfl_xor(s, 1, 16);
        s += __shfl_xor(s, 2, 16);
        s += __shfl_xor(s, 4, 16);
        s += __shfl_xor(s, 8, 16);
        if (lcol == 0)
            allPath[r0 + mrow + k] = __builtin_fmaf(LN2, __builtin_amdgcn_logf(s), mx);
    }
}

// ---------------------------------------------------------------------------
// Gold path score: one block per sequence, 256 threads reduce over T.
// ---------------------------------------------------------------------------
__global__ __launch_bounds__(256) void crf_gold(
    const float*     __restrict__ logits,
    const long long* __restrict__ tags,     // int64 tags
    const int*       __restrict__ mask,
    const float*     __restrict__ transitions,
    const float*     __restrict__ start_trans,
    const float*     __restrict__ end_trans,
    float* __restrict__ gold)
{
    __shared__ float sacc[256];
    __shared__ int   smsk[256];
    const int b = blockIdx.x, tid = threadIdx.x;

    float acc = 0.f;
    int   msum = 0;
    for (int t = tid; t < T_LEN; t += 256) {
        long base = (long)b * T_LEN + t;
        int tg = (int)tags[base];
        int mk = mask[base];
        float mf = (mk > 0) ? 1.f : 0.f;
        acc += logits[base * KTAG + tg] * mf;                 // emit score (all t)
        if (t > 0) {
            int tp = (int)tags[base - 1];
            acc += transitions[tp * KTAG + tg] * mf;          // trans score (t>=1)
        }
        msum += mk;
    }
    sacc[tid] = acc; smsk[tid] = msum;
    __syncthreads();
    for (int s = 128; s > 0; s >>= 1) {
        if (tid < s) { sacc[tid] += sacc[tid + s]; smsk[tid] += smsk[tid + s]; }
        __syncthreads();
    }
    if (tid == 0) {
        int last = smsk[0] - 1;
        float g = sacc[0]
                + start_trans[(int)tags[(long)b * T_LEN]]
                + end_trans[(int)tags[(long)b * T_LEN + last]];
        gold[b] = g;
    }
}

// ---------------------------------------------------------------------------
// Final: mean(all_path - gold)
// ---------------------------------------------------------------------------
__global__ __launch_bounds__(256) void crf_final(
    const float* __restrict__ allPath,
    const float* __restrict__ gold,
    float* __restrict__ out)
{
    __shared__ float s[256];
    const int tid = threadIdx.x;
    s[tid] = allPath[tid] - gold[tid];
    __syncthreads();
    for (int k = 128; k > 0; k >>= 1) {
        if (tid < k) s[tid] += s[tid + k];
        __syncthreads();
    }
    if (tid == 0) out[0] = s[0] * (1.0f / BATCH);
}

// ---------------------------------------------------------------------------
extern "C" void kernel_launch(void* const* d_in, const int* in_sizes, int n_in,
                              void* d_out, int out_size, void* d_ws, size_t ws_size,
                              hipStream_t stream) {
    const float*     logits      = (const float*)d_in[0];
    const long long* tags        = (const long long*)d_in[1];   // int64
    const int*       mask        = (const int*)d_in[2];
    const float*     transitions = (const float*)d_in[3];
    const float*     start_trans = (const float*)d_in[4];
    const float*     end_trans   = (const float*)d_in[5];

    float* allPath = (float*)d_ws;          // [256]
    float* gold    = allPath + BATCH;       // [256]

    crf_forward_wmma<<<BATCH / 16, 32, 0, stream>>>(
        logits, mask, transitions, start_trans, end_trans, allPath);
    crf_gold<<<BATCH, 256, 0, stream>>>(
        logits, tags, mask, transitions, start_trans, end_trans, gold);
    crf_final<<<1, 256, 0, stream>>>(allPath, gold, (float*)d_out);
}
